// ANA2B_49409303773501
// MI455X (gfx1250) — matrix operand order repaired and checked by reading, not compiled, not run on hardware
//
#include <hip/hip_runtime.h>

// ---------------------------------------------------------------------------
// CDNA5 / gfx1250 implementation.
// Heavy phase: per-pair MLPs done as 16-row WMMA tiles with
// v_wmma_f32_16x16x32_f16 (f16 in, f32 accum). One wave32 per 16-pair tile.
//
// d_in flattening assumption (JAX tree_flatten: dict keys sorted):
//  0..17  tensors in setup_inputs order up to indices_j
//  18..23 params['K']  : W0,b0,W1,b1,W2,b2
//  24..29 params['S']  : W0,b0,W1,b1,W2,b2
//  30..35 params['femb']: W0,b0,W1,b1,W2,b2
//  36..43 params['gnn'] : embed_W, embed_b, msg_W1, msg_W2, msg_b1, msg_b2,
//                         upd_W, upd_b   (ASCII sort: 'W' < 'b')
//  44     batch_size
// ---------------------------------------------------------------------------

typedef __attribute__((ext_vector_type(16))) _Float16 v16h;
typedef __attribute__((ext_vector_type(8)))  float    v8f;

#define SB 160      // LDS activation row stride (f16 elements), >= max padded K
#define NNODE 384
#define WB() __builtin_amdgcn_wave_barrier()

// cheap swish: x * sigmoid(x), sigmoid via v_rcp_f32 (activations feed f16 WMMA
// inputs, so one-ulp rcp is ample precision and avoids the precise-div expansion)
__device__ __forceinline__ float swishf(float x) {
  return x * __builtin_amdgcn_rcpf(1.0f + __expf(-x));
}
__device__ __forceinline__ float softplusf(float x) {
  return fmaxf(x, 0.0f) + __logf(1.0f + __expf(-fabsf(x)));
}

// ------------------------------- GNN (small) --------------------------------

__global__ void zerof(float* p, int n) {
  int t = blockIdx.x * blockDim.x + threadIdx.x;
  if (t < n) p[t] = 0.0f;
}

__global__ void gnn_embed(const float* __restrict__ nodes,
                          const float* __restrict__ W, const float* __restrict__ b,
                          float* __restrict__ h, int N) {
  int t = blockIdx.x * blockDim.x + threadIdx.x;
  if (t >= N * 64) return;
  int n = t >> 6, o = t & 63;
  float acc = b[o];
  for (int k = 0; k < 32; ++k) acc += nodes[n * 32 + k] * W[k * 64 + o];
  h[t] = swishf(acc);
}

__global__ void gnn_msg1(const float* __restrict__ h, const float* __restrict__ ef,
                         const int* __restrict__ esrc,
                         const float* __restrict__ W, const float* __restrict__ b,
                         float* __restrict__ m1, int E) {
  int t = blockIdx.x * blockDim.x + threadIdx.x;
  if (t >= E * 64) return;
  int e = t >> 6, o = t & 63;
  int src = esrc[e];
  float acc = b[o];
  for (int k = 0; k < 64; ++k) acc += h[src * 64 + k] * W[k * 64 + o];
  for (int k = 0; k < 8; ++k)  acc += ef[e * 8 + k] * W[(64 + k) * 64 + o];
  m1[t] = swishf(acc);
}

__global__ void gnn_msg2(const float* __restrict__ m1, const int* __restrict__ edst,
                         const float* __restrict__ W, const float* __restrict__ b,
                         float* __restrict__ agg, int E) {
  int t = blockIdx.x * blockDim.x + threadIdx.x;
  if (t >= E * 64) return;
  int e = t >> 6, o = t & 63;
  float acc = b[o];
  for (int k = 0; k < 64; ++k) acc += m1[e * 64 + k] * W[k * 64 + o];
  atomicAdd(&agg[edst[e] * 64 + o], swishf(acc));
}

__global__ void gnn_upd(const float* __restrict__ hpre, const float* __restrict__ agg,
                        const float* __restrict__ W, const float* __restrict__ b,
                        float* __restrict__ h, int N) {
  int t = blockIdx.x * blockDim.x + threadIdx.x;
  if (t >= N * 64) return;
  int n = t >> 6, o = t & 63;
  float acc = b[o];
  for (int k = 0; k < 64; ++k) acc += hpre[n * 64 + k] * W[k * 64 + o];
  for (int k = 0; k < 64; ++k) acc += agg[n * 64 + k]  * W[(64 + k) * 64 + o];
  h[t] = swishf(acc);
}

// ------------------------- weight packing (f32 -> f16 B-frags) --------------
// Packed layout: dst[((kb*nnb+nb)*32 + lane)*16 + e] = W[kb*32 + kh + e][nb*16 + (lane&15)]
// where kh = (lane&16) ? 16 : 0.  Zero-pad out of range.

__global__ void pack_w(const float* __restrict__ W, _Float16* __restrict__ dst,
                       int inF, int outF, int inP, int outP) {
  int nnb = outP >> 4;
  int total = (inP >> 5) * nnb * 32 * 16;
  int t = blockIdx.x * blockDim.x + threadIdx.x;
  if (t >= total) return;
  int e = t & 15;
  int lane = (t >> 4) & 31;
  int rest = t >> 9;
  int nb = rest % nnb;
  int kb = rest / nnb;
  int kh = (lane & 16) ? 16 : 0;
  int r = kb * 32 + kh + e;
  int c = nb * 16 + (lane & 15);
  float v = (r < inF && c < outF) ? W[r * outF + c] : 0.0f;
  dst[t] = (_Float16)v;
}

// ------------------------------- pair kernel --------------------------------

struct PairArgs {
  const float *coords1, *coords2, *dm, *mono1, *mono2, *dip1, *dip2, *quad1, *quad2;
  const int *ib, *ii, *ij;
  const float *h1, *h2;
  const _Float16 *Fp0, *Fp1, *Fp2, *Sp0, *Sp1, *Sp2, *Kp0, *Kp1, *Kp2;
  const float *Fb0, *Fb1, *Fb2, *Sb0, *Sb1, *Sb2, *Kb0, *Kb1, *Kb2;
  float* out;
  int P;
};

struct WaveLDS {
  _Float16 buf0[16 * SB];
  _Float16 buf1[16 * SB];
  _Float16 buf2[16 * SB];
  float fsum[16 * 128];      // femb accumulator; reused as raw final-layer out
  float G[16 * 10];
  float dist[16 * 5];
  float sw[16], r1v[16], r2v[16];
  int   bidx[16], si[16], sj[16];
  float s2s[16], s2a[16];
};

// 16-bit A 16x32, contiguous k order (matches pack_w): lane<16: K 0..15, lane>=16: K 16..31.
__device__ __forceinline__ v16h load_a(const _Float16* __restrict__ buf, int kb, int lane) {
  const int row = lane & 15;
  const _Float16* p = buf + row * SB + kb * 32 + ((lane & 16) ? 16 : 0);
  v16h a;
#pragma unroll
  for (int e = 0; e < 8; ++e) { a[e] = p[e]; a[e + 8] = p[8 + e]; }
  return a;
}

// modes: 0 = swish -> f16 outBuf ; 1 = identity, accBuf += ; 2 = identity, accBuf =
__device__ __forceinline__ void mlp_layer(
    const _Float16* __restrict__ inBuf, int inP,
    const _Float16* __restrict__ wp, const float* __restrict__ bias,
    int outF, int outP,
    _Float16* __restrict__ outBuf,
    float* __restrict__ accBuf, int accStride,
    int mode, int lane) {
  WB();
  const int nkb = inP >> 5, nnb = outP >> 4;
  const int col0 = lane & 15;
  const int rowh = (lane & 16) ? 8 : 0;
  for (int nb = 0; nb < nnb; ++nb) {
    v8f c = {0.f, 0.f, 0.f, 0.f, 0.f, 0.f, 0.f, 0.f};
    for (int kb = 0; kb < nkb; ++kb) {
      v16h a = load_a(inBuf, kb, lane);
      v16h b = *(const v16h*)(wp + (((size_t)(kb * nnb + nb) * 32 + lane) << 4));
      c = __builtin_amdgcn_wmma_f32_16x16x32_f16(
              /*neg_a=*/false, a, /*neg_b=*/false, b,
              /*c_mod=*/(short)0, c, /*reuse_a=*/false, /*reuse_b=*/false);
    }
    const int col = nb * 16 + col0;
    const float bv = (col < outF) ? bias[col] : 0.0f;
#pragma unroll
    for (int v = 0; v < 8; ++v) {
      const int row = v + rowh;
      float x = c[v] + bv;
      if (mode == 0)      outBuf[row * SB + col] = (_Float16)swishf(x);
      else if (mode == 1) accBuf[row * accStride + col] += x;
      else                accBuf[row * accStride + col] = x;
    }
  }
  WB();
}

__launch_bounds__(64)
__global__ void pair_kernel(PairArgs a) {
  __shared__ WaveLDS lds[2];
  const int lane = threadIdx.x & 31;
  const int wv = threadIdx.x >> 5;
  WaveLDS& w = lds[wv];
  const int tile = blockIdx.x * 2 + wv;
  const int p0 = tile * 16;
  if (p0 >= a.P) return;   // wave-uniform

  // ---- phase 1: geometry, multipole G, switch, gaussians (lanes 0..15) ----
  if (lane < 16) {
    const int p = p0 + lane;
    const bool ok = p < a.P;
    const int b = ok ? a.ib[p] : 0;
    const int i = ok ? a.ii[p] : 0;
    const int j = ok ? a.ij[p] : 0;
    w.bidx[lane] = b; w.si[lane] = i; w.sj[lane] = j;
    float R1 = ok ? a.dm[((size_t)b * NNODE + i) * NNODE + j] : 1.0f;
    float R2 = R1 * R1;
    const float* c1 = a.coords1 + ((size_t)b * NNODE + i) * 3;
    const float* c2 = a.coords2 + ((size_t)b * NNODE + j) * 3;
    float rx[3];
#pragma unroll
    for (int k = 0; k < 3; ++k) rx[k] = (c2[k] - c1[k]) / R1;
    const float m1 = a.mono1[(size_t)b * NNODE + i];
    const float m2 = a.mono2[(size_t)b * NNODE + j];
    const float* d1 = a.dip1 + ((size_t)b * NNODE + i) * 3;
    const float* d2 = a.dip2 + ((size_t)b * NNODE + j) * 3;
    const float* q1 = a.quad1 + ((size_t)b * NNODE + i) * 9;
    const float* q2 = a.quad2 + ((size_t)b * NNODE + j) * 9;
    float D1 = 0, D2 = 0, dd = 0, qq = 0;
    float Q1R[3] = {0, 0, 0}, Q2R[3] = {0, 0, 0};
#pragma unroll
    for (int k = 0; k < 3; ++k) { D1 += d1[k] * rx[k]; D2 += d2[k] * rx[k]; dd += d1[k] * d2[k]; }
#pragma unroll
    for (int k = 0; k < 3; ++k)
#pragma unroll
      for (int l = 0; l < 3; ++l) {
        Q1R[k] += q1[k * 3 + l] * rx[l];
        Q2R[k] += q2[k * 3 + l] * rx[l];
        qq += q1[k * 3 + l] * q2[k * 3 + l];
      }
    float Q1RR = 0, Q2RR = 0, qd2 = 0, qd1 = 0, qrr = 0;
#pragma unroll
    for (int k = 0; k < 3; ++k) {
      Q1RR += Q1R[k] * rx[k]; Q2RR += Q2R[k] * rx[k];
      qd2 += Q1R[k] * d2[k];  qd1 += Q2R[k] * d1[k];  qrr += Q1R[k] * Q2R[k];
    }
    float* G = w.G + lane * 10;
    G[0] = m1 * m2;
    G[1] = D1 * m2 - D2 * m1;
    G[2] = dd;
    G[3] = -D1 * D2;
    G[4] = 2.0f * (qd2 - qd1);
    G[5] = Q1RR * m2 + Q2RR * m1;
    G[6] = 2.0f * qq;
    G[7] = -4.0f * qrr;
    G[8] = -Q1RR * D2 + Q2RR * D1;
    G[9] = Q1RR * Q2RR;
    float x = fminf(fmaxf(R1 - 9.0f, 0.0f), 1.0f);            // (R-r0)/(r1-r0), r0=9,r1=10
    float sw = 1.0f - x * x * x * (x * (6.0f * x - 15.0f) + 10.0f);
    if (!ok) sw = 0.0f;
    w.sw[lane] = sw; w.r1v[lane] = R1; w.r2v[lane] = R2;
    const float gam[5] = {0.1f, 0.17782794f, 0.31622776f, 0.56234133f, 1.0f};
#pragma unroll
    for (int g = 0; g < 5; ++g) w.dist[lane * 5 + g] = __expf(-gam[g] * R2) * sw;
  }
  WB();

  // ---- gather node features (both concat orders, f16) ----
  for (int t = lane; t < 16 * 128; t += 32) {
    int r = t >> 7, c = t & 127;
    float v = (c < 64) ? a.h1[w.si[r] * 64 + c] : a.h2[w.sj[r] * 64 + (c - 64)];
    _Float16 hv = (_Float16)v;
    w.buf0[r * SB + c] = hv;          // [nf1, nf2]
    w.buf2[r * SB + (c ^ 64)] = hv;   // [nf2, nf1]
  }
  WB();

  // ---- femb, symmetric sum (identity final act) ----
  mlp_layer(w.buf0, 128, a.Fp0, a.Fb0, 128, 128, w.buf1, nullptr, 0, 0, lane);
  mlp_layer(w.buf1, 128, a.Fp1, a.Fb1, 128, 128, w.buf0, nullptr, 0, 0, lane);
  mlp_layer(w.buf0, 128, a.Fp2, a.Fb2, 128, 128, nullptr, w.fsum, 128, 2, lane);
  mlp_layer(w.buf2, 128, a.Fp0, a.Fb0, 128, 128, w.buf1, nullptr, 0, 0, lane);
  mlp_layer(w.buf1, 128, a.Fp1, a.Fb1, 128, 128, w.buf2, nullptr, 0, 0, lane);
  mlp_layer(w.buf2, 128, a.Fp2, a.Fb2, 128, 128, nullptr, w.fsum, 128, 1, lane);

  // ---- build S input [G | nodefeat | dist | 0] and K input [nodefeat | dist | 0] ----
  for (int t = lane; t < 16 * SB; t += 32) {
    int r = t / SB, c = t - r * SB;
    float vS, vK;
    if (c < 10)        vS = w.G[r * 10 + c];
    else if (c < 138)  vS = w.fsum[r * 128 + (c - 10)];
    else if (c < 143)  vS = w.dist[r * 5 + (c - 138)];
    else               vS = 0.0f;
    if (c < 128)       vK = w.fsum[r * 128 + c];
    else if (c < 133)  vK = w.dist[r * 5 + (c - 128)];
    else               vK = 0.0f;
    w.buf1[r * SB + c] = (_Float16)vS;
    w.buf2[r * SB + c] = (_Float16)vK;
  }
  WB();

  // ---- S MLP: 160->128->128->2 (softplus * sw) ----
  mlp_layer(w.buf1, 160, a.Sp0, a.Sb0, 128, 128, w.buf0, nullptr, 0, 0, lane);
  mlp_layer(w.buf0, 128, a.Sp1, a.Sb1, 128, 128, w.buf1, nullptr, 0, 0, lane);
  mlp_layer(w.buf1, 128, a.Sp2, a.Sb2, 2, 16, nullptr, w.fsum, 16, 2, lane);
  if (lane < 16) {
    w.s2s[lane] = softplusf(w.fsum[lane * 16 + 0]) * w.sw[lane];
    w.s2a[lane] = softplusf(w.fsum[lane * 16 + 1]) * w.sw[lane];
  }
  WB();

  // ---- K MLP: 160->128->128->3 (softplus) ----
  mlp_layer(w.buf2, 160, a.Kp0, a.Kb0, 128, 128, w.buf0, nullptr, 0, 0, lane);
  mlp_layer(w.buf0, 128, a.Kp1, a.Kb1, 128, 128, w.buf2, nullptr, 0, 0, lane);
  mlp_layer(w.buf2, 128, a.Kp2, a.Kb2, 3, 16, nullptr, w.fsum + 256, 16, 2, lane);
  WB();

  // ---- V = K1*S2s/R1 + K2*S2s/R2 - Kat*S2a ; segment-sum over batch ----
  if (lane < 16 && (p0 + lane) < a.P) {
    float K1 = softplusf(w.fsum[256 + lane * 16 + 0]);
    float K2 = softplusf(w.fsum[256 + lane * 16 + 1]);
    float Ka = softplusf(w.fsum[256 + lane * 16 + 2]);
    float V = K1 * w.s2s[lane] / w.r1v[lane] + K2 * w.s2s[lane] / w.r2v[lane]
              - Ka * w.s2a[lane];
    atomicAdd(a.out + w.bidx[lane], V);
  }
}

// ------------------------------- launch -------------------------------------

extern "C" void kernel_launch(void* const* d_in, const int* in_sizes, int n_in,
                              void* d_out, int out_size, void* d_ws, size_t ws_size,
                              hipStream_t stream) {
  (void)n_in; (void)ws_size;
  const float* coords1 = (const float*)d_in[0];
  const float* coords2 = (const float*)d_in[1];
  const float* dm      = (const float*)d_in[2];
  const float* mono1   = (const float*)d_in[3];
  const float* mono2   = (const float*)d_in[4];
  const float* dip1    = (const float*)d_in[5];
  const float* dip2    = (const float*)d_in[6];
  const float* quad1   = (const float*)d_in[7];
  const float* quad2   = (const float*)d_in[8];
  const float* nodes1  = (const float*)d_in[9];
  const float* nodes2  = (const float*)d_in[10];
  const float* ef1     = (const float*)d_in[11];
  const float* ef2     = (const float*)d_in[12];
  const int*   eidx1   = (const int*)d_in[13];
  const int*   eidx2   = (const int*)d_in[14];
  const int*   ib      = (const int*)d_in[15];
  const int*   ii      = (const int*)d_in[16];
  const int*   ij      = (const int*)d_in[17];
  const float* K_W0 = (const float*)d_in[18]; const float* K_b0 = (const float*)d_in[19];
  const float* K_W1 = (const float*)d_in[20]; const float* K_b1 = (const float*)d_in[21];
  const float* K_W2 = (const float*)d_in[22]; const float* K_b2 = (const float*)d_in[23];
  const float* S_W0 = (const float*)d_in[24]; const float* S_b0 = (const float*)d_in[25];
  const float* S_W1 = (const float*)d_in[26]; const float* S_b1 = (const float*)d_in[27];
  const float* S_W2 = (const float*)d_in[28]; const float* S_b2 = (const float*)d_in[29];
  const float* F_W0 = (const float*)d_in[30]; const float* F_b0 = (const float*)d_in[31];
  const float* F_W1 = (const float*)d_in[32]; const float* F_b1 = (const float*)d_in[33];
  const float* F_W2 = (const float*)d_in[34]; const float* F_b2 = (const float*)d_in[35];
  const float* embW = (const float*)d_in[36]; const float* embB = (const float*)d_in[37];
  const float* mW1  = (const float*)d_in[38]; const float* mW2  = (const float*)d_in[39];
  const float* mB1  = (const float*)d_in[40]; const float* mB2  = (const float*)d_in[41];
  const float* uW   = (const float*)d_in[42]; const float* uB   = (const float*)d_in[43];

  const int P = in_sizes[15];
  const int N = NNODE, E = 1024;

  char* base = (char*)d_ws;
  size_t off = 0;
  auto carve = [&](size_t bytes) -> void* {
    void* p = base + off;
    off += (bytes + 255) & ~(size_t)255;
    return p;
  };
  float* h1p  = (float*)carve((size_t)N * 64 * 4);
  float* h2p  = (float*)carve((size_t)N * 64 * 4);
  float* agg1 = (float*)carve((size_t)N * 64 * 4);
  float* agg2 = (float*)carve((size_t)N * 64 * 4);
  float* h1   = (float*)carve((size_t)N * 64 * 4);
  float* h2   = (float*)carve((size_t)N * 64 * 4);
  float* m1a  = (float*)carve((size_t)E * 64 * 4);
  float* m1b  = (float*)carve((size_t)E * 64 * 4);
  _Float16* Fp0 = (_Float16*)carve(128 * 128 * 2);
  _Float16* Fp1 = (_Float16*)carve(128 * 128 * 2);
  _Float16* Fp2 = (_Float16*)carve(128 * 128 * 2);
  _Float16* Sp0 = (_Float16*)carve(160 * 128 * 2);
  _Float16* Sp1 = (_Float16*)carve(128 * 128 * 2);
  _Float16* Sp2 = (_Float16*)carve(128 * 16 * 2);
  _Float16* Kp0 = (_Float16*)carve(160 * 128 * 2);
  _Float16* Kp1 = (_Float16*)carve(128 * 128 * 2);
  _Float16* Kp2 = (_Float16*)carve(128 * 16 * 2);

  // init accumulators
  zerof<<<(N * 64 + 255) / 256, 256, 0, stream>>>(agg1, N * 64);
  zerof<<<(N * 64 + 255) / 256, 256, 0, stream>>>(agg2, N * 64);
  zerof<<<1, 64, 0, stream>>>((float*)d_out, out_size);

  // GNN (tiny)
  gnn_embed<<<(N * 64 + 255) / 256, 256, 0, stream>>>(nodes1, embW, embB, h1p, N);
  gnn_embed<<<(N * 64 + 255) / 256, 256, 0, stream>>>(nodes2, embW, embB, h2p, N);
  gnn_msg1<<<(E * 64 + 255) / 256, 256, 0, stream>>>(h1p, ef1, eidx1, mW1, mB1, m1a, E);
  gnn_msg1<<<(E * 64 + 255) / 256, 256, 0, stream>>>(h2p, ef2, eidx2, mW1, mB1, m1b, E);
  gnn_msg2<<<(E * 64 + 255) / 256, 256, 0, stream>>>(m1a, eidx1 + E, mW2, mB2, agg1, E);
  gnn_msg2<<<(E * 64 + 255) / 256, 256, 0, stream>>>(m1b, eidx2 + E, mW2, mB2, agg2, E);
  gnn_upd<<<(N * 64 + 255) / 256, 256, 0, stream>>>(h1p, agg1, uW, uB, h1, N);
  gnn_upd<<<(N * 64 + 255) / 256, 256, 0, stream>>>(h2p, agg2, uW, uB, h2, N);

  // pack weights to f16 fragments
  auto pack = [&](const float* W, _Float16* dst, int inF, int outF, int inP, int outP) {
    int total = (inP >> 5) * (outP >> 4) * 512;
    pack_w<<<(total + 255) / 256, 256, 0, stream>>>(W, dst, inF, outF, inP, outP);
  };
  pack(F_W0, Fp0, 128, 128, 128, 128);
  pack(F_W1, Fp1, 128, 128, 128, 128);
  pack(F_W2, Fp2, 128, 128, 128, 128);
  pack(S_W0, Sp0, 143, 128, 160, 128);
  pack(S_W1, Sp1, 128, 128, 128, 128);
  pack(S_W2, Sp2, 128,   2, 128,  16);
  pack(K_W0, Kp0, 133, 128, 160, 128);
  pack(K_W1, Kp1, 128, 128, 128, 128);
  pack(K_W2, Kp2, 128,   3, 128,  16);

  // main WMMA pair kernel: 2 waves / block, 16 pairs / wave
  PairArgs pa;
  pa.coords1 = coords1; pa.coords2 = coords2; pa.dm = dm;
  pa.mono1 = mono1; pa.mono2 = mono2; pa.dip1 = dip1; pa.dip2 = dip2;
  pa.quad1 = quad1; pa.quad2 = quad2;
  pa.ib = ib; pa.ii = ii; pa.ij = ij;
  pa.h1 = h1; pa.h2 = h2;
  pa.Fp0 = Fp0; pa.Fp1 = Fp1; pa.Fp2 = Fp2;
  pa.Sp0 = Sp0; pa.Sp1 = Sp1; pa.Sp2 = Sp2;
  pa.Kp0 = Kp0; pa.Kp1 = Kp1; pa.Kp2 = Kp2;
  pa.Fb0 = F_b0; pa.Fb1 = F_b1; pa.Fb2 = F_b2;
  pa.Sb0 = S_b0; pa.Sb1 = S_b1; pa.Sb2 = S_b2;
  pa.Kb0 = K_b0; pa.Kb1 = K_b1; pa.Kb2 = K_b2;
  pa.out = (float*)d_out; pa.P = P;

  int nTiles = (P + 15) / 16;
  int grid = (nTiles + 1) / 2;
  pair_kernel<<<grid, 64, 0, stream>>>(pa);
}
